// CC_module_21285857919587
// MI455X (gfx1250) — matrix operand order repaired and checked by reading, not compiled, not run on hardware
//
#include <hip/hip_runtime.h>
#include <math.h>

typedef _Float16 half_t;
typedef __attribute__((ext_vector_type(16))) _Float16 v16h;
typedef __attribute__((ext_vector_type(8)))  _Float16 v8h;
typedef __attribute__((ext_vector_type(4)))  _Float16 v4h;
typedef __attribute__((ext_vector_type(8)))  float    v8f;
typedef __attribute__((ext_vector_type(4)))  float    v4f;

#define BB 8
#define CC 512
#define CQ 64
#define HH 64
#define WW 64
#define HWP 4096   // H*W

union V16U { v16h v; v8h h[2]; };

// ---- WMMA fragment helpers (wave32, 16x16x32 f16) ----
// A 16x32 (MxK), row-major source with row stride (halves).
// lane = m + 16*half ; elements 0..7 -> k = half*8+0..7 ; 8..15 -> k = 16+half*8+0..7
__device__ __forceinline__ v16h load_frag_a(const half_t* base, int row_stride) {
    int lane = threadIdx.x & 31;
    int hf = lane >> 4;
    int m  = lane & 15;
    const half_t* p = base + m * row_stride + hf * 8;
    V16U r;
    r.h[0] = *(const v8h*)(p);
    r.h[1] = *(const v8h*)(p + 16);
    return r.v;
}

// B 32x16 (KxN), source stored TRANSPOSED as [n][k] with row stride (halves).
// lane = n + 16*khalf ; elements 0..7 -> k = khalf*8+0..7 ; 8..15 -> k = 16+khalf*8+0..7
__device__ __forceinline__ v16h load_frag_bT(const half_t* base, int row_stride) {
    int lane = threadIdx.x & 31;
    int kh = lane >> 4;
    int n  = lane & 15;
    const half_t* p = base + n * row_stride + kh * 8;
    V16U r;
    r.h[0] = *(const v8h*)(p);
    r.h[1] = *(const v8h*)(p + 16);
    return r.v;
}

// A 16x32 with arbitrary element strides (for column-direction v reads)
__device__ __forceinline__ v16h load_frag_a_strided(const half_t* base, int stride_m, int stride_k) {
    int lane = threadIdx.x & 31;
    int hf = lane >> 4;
    int m  = lane & 15;
    v16h r;
#pragma unroll
    for (int j = 0; j < 16; ++j) {
        int k = ((j >> 3) << 4) + hf * 8 + (j & 7);
        r[j] = base[m * stride_m + k * stride_k];
    }
    return r;
}

__device__ __forceinline__ v8f wmma16(v16h a, v16h b, v8f c) {
    return __builtin_amdgcn_wmma_f32_16x16x32_f16(false, a, false, b, (short)0, c, false, false);
}

// =====================================================================
// Kernel 1: 1x1 conv as GEMM.  Y[b,o,p] = sum_c Wt[o,c]*X[b,c,p] + bias[o]
// grid (HWP/128, O/64, B), block 256 (8 waves). Output f16.
// =====================================================================
__global__ void proj_gemm(const float* __restrict__ X, const float* __restrict__ Wt,
                          const float* __restrict__ bias, half_t* __restrict__ Y, int O) {
    __shared__ __align__(16) half_t As[64 * 40];   // [o][c] stride 40
    __shared__ __align__(16) half_t Bs[128 * 40];  // [p][c] stride 40 (transposed)
    int b  = blockIdx.z;
    int m0 = blockIdx.y * 64;
    int n0 = blockIdx.x * 128;
    int tid = threadIdx.x;
    int wave = tid >> 5, lane = tid & 31;
    int wm = (wave >> 1) * 16;   // 4 waves along M
    int wn = (wave & 1) * 64;    // 2 waves along N, 4 tiles each

    const float* Xb = X + (size_t)b * CC * HWP;
    v8f acc[4] = {v8f{}, v8f{}, v8f{}, v8f{}};

    for (int k0 = 0; k0 < CC; k0 += 32) {
        // ---- phase 1: gather tiles into registers (many outstanding loads) ----
        v4f ta[2];  // A: 64x32 f32 = 512 float4, 2 per thread
#pragma unroll
        for (int t = 0; t < 2; ++t) {
            int iv = tid + t * 256;          // 0..511
            int r  = iv >> 3;                // 8 float4 per 32-wide row
            int c4 = iv & 7;
            ta[t] = *(const v4f*)&Wt[(size_t)(m0 + r) * CC + k0 + c4 * 4];
        }
        v4f tb[4];  // B: 32x128 f32 = 1024 float4, 4 per thread
#pragma unroll
        for (int t = 0; t < 4; ++t) {
            int iv = tid + t * 256;          // 0..1023
            int c  = iv >> 5;                // 32 float4 per 128-wide row
            int p4 = iv & 31;
            tb[t] = *(const v4f*)&Xb[(size_t)(k0 + c) * HWP + n0 + p4 * 4];
        }
        // prefetch next K-tile of X into cache while this tile computes
        if (k0 + 32 < CC) {
#pragma unroll
            for (int t = 0; t < 4; ++t) {
                int iv = tid + t * 256;
                int c  = iv >> 5;
                int p4 = iv & 31;
                __builtin_prefetch(&Xb[(size_t)(k0 + 32 + c) * HWP + n0 + p4 * 4], 0, 3);
            }
        }
        __syncthreads();   // previous iteration's LDS reads complete
        // ---- phase 2: store to LDS ----
#pragma unroll
        for (int t = 0; t < 2; ++t) {
            int iv = tid + t * 256;
            int r  = iv >> 3;
            int c4 = iv & 7;
            v4h h;
#pragma unroll
            for (int u = 0; u < 4; ++u) h[u] = (half_t)ta[t][u];
            *(v4h*)&As[r * 40 + c4 * 4] = h;
        }
#pragma unroll
        for (int t = 0; t < 4; ++t) {
            int iv = tid + t * 256;
            int c  = iv >> 5;
            int p4 = iv & 31;
#pragma unroll
            for (int u = 0; u < 4; ++u) Bs[(p4 * 4 + u) * 40 + c] = (half_t)tb[t][u];
        }
        __syncthreads();
        // ---- phase 3: WMMA ----
        v16h a = load_frag_a(&As[wm * 40], 40);
#pragma unroll
        for (int j = 0; j < 4; ++j) {
            v16h bf = load_frag_bT(&Bs[(wn + j * 16) * 40], 40);
            acc[j] = wmma16(a, bf, acc[j]);
        }
    }
    int mloc = (lane >> 4) * 8, n = lane & 15;
    float bi[8];
#pragma unroll
    for (int r = 0; r < 8; ++r) bi[r] = bias[m0 + wm + mloc + r];
#pragma unroll
    for (int j = 0; j < 4; ++j) {
#pragma unroll
        for (int r = 0; r < 8; ++r) {
            int m = m0 + wm + mloc + r;
            int p = n0 + wn + j * 16 + n;
            Y[((size_t)b * O + m) * HWP + p] = (half_t)(acc[j][r] + bi[r]);
        }
    }
}

// =====================================================================
// Kernel 2: energies. mode 0: E_H[b,h,w,H'] (fixed w) ; mode 1: E_W (fixed h)
// grid (64, B), block 256.  E layout: [b][h][w][128] f32 (H part 0..63, W part 64..127)
// =====================================================================
__global__ void energy_kernel(const half_t* __restrict__ Q, const half_t* __restrict__ Kt,
                              float* __restrict__ E, int mode) {
    __shared__ __align__(16) half_t Aq[64 * 72]; // [m][c]
    __shared__ __align__(16) half_t Bk[64 * 72]; // [n][c]
    int fix = blockIdx.x;
    int b   = blockIdx.y;
    int tid = threadIdx.x, wave = tid >> 5, lane = tid & 31;
    const half_t* Qb = Q  + (size_t)b * CQ * HWP;
    const half_t* Kb = Kt + (size_t)b * CQ * HWP;
    // two-phase staging, pixel coordinate fastest (mode 1 rows are contiguous)
    half_t tq[16], tk[16];
#pragma unroll
    for (int t = 0; t < 16; ++t) {
        int i = tid + t * 256;
        int m = i & 63, c = i >> 6;
        int pix = mode ? (fix * WW + m) : (m * WW + fix);
        tq[t] = Qb[(size_t)c * HWP + pix];
        tk[t] = Kb[(size_t)c * HWP + pix];
    }
#pragma unroll
    for (int t = 0; t < 16; ++t) {
        int i = tid + t * 256;
        int m = i & 63, c = i >> 6;
        Aq[m * 72 + c] = tq[t];
        Bk[m * 72 + c] = tk[t];
    }
    __syncthreads();
    int mi = wave >> 1;             // 4 strips along rows
    int nb = (wave & 1) * 2;        // 2 n-tiles each
    v8f acc[2] = {v8f{}, v8f{}};
#pragma unroll
    for (int k0 = 0; k0 < CQ; k0 += 32) {
        v16h a = load_frag_a(&Aq[mi * 16 * 72 + k0], 72);
#pragma unroll
        for (int j = 0; j < 2; ++j) {
            v16h bf = load_frag_bT(&Bk[(nb + j) * 16 * 72 + k0], 72);
            acc[j] = wmma16(a, bf, acc[j]);
        }
    }
    int mloc = (lane >> 4) * 8, n = lane & 15;
    int cbase = mode ? 64 : 0;
#pragma unroll
    for (int j = 0; j < 2; ++j) {
#pragma unroll
        for (int r = 0; r < 8; ++r) {
            int m   = mi * 16 + mloc + r;
            int col = (nb + j) * 16 + n;
            int h = mode ? fix : m;
            int w = mode ? m : fix;
            E[(((size_t)b * HH + h) * WW + w) * 128 + cbase + col] = acc[j][r];
        }
    }
}

// =====================================================================
// Kernel 3: softmax over 128 logits per pixel -> f16 attention.
// one wave per pixel; block = 8 pixels.
// =====================================================================
__global__ void softmax_kernel(const float* __restrict__ E, half_t* __restrict__ A) {
    int wave = threadIdx.x >> 5, lane = threadIdx.x & 31;
    size_t pix = (size_t)blockIdx.x * 8 + wave;
    v4f v = *(const v4f*)(E + pix * 128 + lane * 4);
    float mx = fmaxf(fmaxf(v[0], v[1]), fmaxf(v[2], v[3]));
#pragma unroll
    for (int off = 16; off; off >>= 1) mx = fmaxf(mx, __shfl_xor(mx, off, 32));
    float s = 0.f;
#pragma unroll
    for (int i = 0; i < 4; ++i) { v[i] = __expf(v[i] - mx); s += v[i]; }
#pragma unroll
    for (int off = 16; off; off >>= 1) s += __shfl_xor(s, off, 32);
    float inv = 1.0f / s;
    v4h h;
#pragma unroll
    for (int i = 0; i < 4; ++i) h[i] = (half_t)(v[i] * inv);
    *(v4h*)(A + pix * 128 + lane * 4) = h;
}

// =====================================================================
// Kernel 4a: row aggregation, writes y = gamma*out_W + x for both streams.
// grid (H, C/128, B), block 256 (8 waves, 16 c-rows each).
// =====================================================================
__global__ void agg_row_kernel(const half_t* __restrict__ V2, const half_t* __restrict__ V1,
                               const half_t* __restrict__ Att,
                               const float* __restrict__ X2, const float* __restrict__ X1,
                               const float* __restrict__ gamma, float* __restrict__ Y) {
    __shared__ __align__(16) half_t Bs[64 * 72];  // [w][W']
    int h  = blockIdx.x;
    int c0 = blockIdx.y * 128;
    int b  = blockIdx.z;
    int tid = threadIdx.x, wave = tid >> 5, lane = tid & 31;
    const half_t* att = Att + (((size_t)b * HH + h) * WW) * 128;
    // vectorized two-phase staging: 64x64 halves = 512 v8h, 2 per thread
    v8h tbv[2];
#pragma unroll
    for (int t = 0; t < 2; ++t) {
        int iv = tid + t * 256;
        int w = iv >> 3, k8 = iv & 7;
        tbv[t] = *(const v8h*)&att[(size_t)w * 128 + 64 + k8 * 8];
    }
#pragma unroll
    for (int t = 0; t < 2; ++t) {
        int iv = tid + t * 256;
        int w = iv >> 3, k8 = iv & 7;
        *(v8h*)&Bs[w * 72 + k8 * 8] = tbv[t];
    }
    __syncthreads();
    float g = gamma[0];
    const half_t* v2row = V2 + ((size_t)b * CC + c0 + wave * 16) * HWP + (size_t)h * WW;
    const half_t* v1row = V1 + ((size_t)b * CC + c0 + wave * 16) * HWP + (size_t)h * WW;
    v8f a2[4] = {v8f{}, v8f{}, v8f{}, v8f{}};
    v8f a1[4] = {v8f{}, v8f{}, v8f{}, v8f{}};
#pragma unroll
    for (int k0 = 0; k0 < 64; k0 += 32) {
        v16h fa2 = load_frag_a(v2row + k0, HWP);  // contiguous k along W'
        v16h fa1 = load_frag_a(v1row + k0, HWP);
#pragma unroll
        for (int j = 0; j < 4; ++j) {
            v16h bf = load_frag_bT(&Bs[j * 16 * 72 + k0], 72);
            a2[j] = wmma16(fa2, bf, a2[j]);
            a1[j] = wmma16(fa1, bf, a1[j]);
        }
    }
    int mloc = (lane >> 4) * 8, n = lane & 15;
    const size_t NTOT = (size_t)BB * CC * HWP;
#pragma unroll
    for (int j = 0; j < 4; ++j) {
#pragma unroll
        for (int r = 0; r < 8; ++r) {
            int c = c0 + wave * 16 + mloc + r;
            int w = j * 16 + n;
            size_t idx = ((size_t)b * CC + c) * HWP + (size_t)h * WW + w;
            Y[idx]        = g * a2[j][r] + X2[idx];
            Y[NTOT + idx] = g * a1[j][r] + X1[idx];
        }
    }
}

// =====================================================================
// Kernel 4b: column aggregation, y += gamma*out_H.
// grid (W, C/128, B), block 256.
// =====================================================================
__global__ void agg_col_kernel(const half_t* __restrict__ V2, const half_t* __restrict__ V1,
                               const half_t* __restrict__ Att,
                               const float* __restrict__ gamma, float* __restrict__ Y) {
    __shared__ __align__(16) half_t Bs[64 * 72];  // [h][H']
    int w  = blockIdx.x;
    int c0 = blockIdx.y * 128;
    int b  = blockIdx.z;
    int tid = threadIdx.x, wave = tid >> 5, lane = tid & 31;
    const half_t* att = Att + ((size_t)b * HH * WW + w) * 128;
    v8h tbv[2];
#pragma unroll
    for (int t = 0; t < 2; ++t) {
        int iv = tid + t * 256;
        int hrow = iv >> 3, k8 = iv & 7;
        tbv[t] = *(const v8h*)&att[(size_t)hrow * WW * 128 + k8 * 8];
    }
#pragma unroll
    for (int t = 0; t < 2; ++t) {
        int iv = tid + t * 256;
        int hrow = iv >> 3, k8 = iv & 7;
        *(v8h*)&Bs[hrow * 72 + k8 * 8] = tbv[t];
    }
    __syncthreads();
    float g = gamma[0];
    const half_t* v2col = V2 + ((size_t)b * CC + c0 + wave * 16) * HWP + w;
    const half_t* v1col = V1 + ((size_t)b * CC + c0 + wave * 16) * HWP + w;
    v8f a2[4] = {v8f{}, v8f{}, v8f{}, v8f{}};
    v8f a1[4] = {v8f{}, v8f{}, v8f{}, v8f{}};
#pragma unroll
    for (int k0 = 0; k0 < 64; k0 += 32) {
        v16h fa2 = load_frag_a_strided(v2col + (size_t)k0 * WW, HWP, WW); // k stride = W
        v16h fa1 = load_frag_a_strided(v1col + (size_t)k0 * WW, HWP, WW);
#pragma unroll
        for (int j = 0; j < 4; ++j) {
            v16h bf = load_frag_bT(&Bs[j * 16 * 72 + k0], 72);
            a2[j] = wmma16(fa2, bf, a2[j]);
            a1[j] = wmma16(fa1, bf, a1[j]);
        }
    }
    int mloc = (lane >> 4) * 8, n = lane & 15;
    const size_t NTOT = (size_t)BB * CC * HWP;
#pragma unroll
    for (int j = 0; j < 4; ++j) {
#pragma unroll
        for (int r = 0; r < 8; ++r) {
            int c = c0 + wave * 16 + mloc + r;
            int h = j * 16 + n;
            size_t idx = ((size_t)b * CC + c) * HWP + (size_t)h * WW + w;
            Y[idx]        += g * a2[j][r];
            Y[NTOT + idx] += g * a1[j][r];
        }
    }
}

// =====================================================================
// Launch
// =====================================================================
extern "C" void kernel_launch(void* const* d_in, const int* in_sizes, int n_in,
                              void* d_out, int out_size, void* d_ws, size_t ws_size,
                              hipStream_t stream) {
    const float* x2    = (const float*)d_in[0];
    const float* x1    = (const float*)d_in[1];
    const float* q_w   = (const float*)d_in[2];
    const float* q_b   = (const float*)d_in[3];
    const float* k_w   = (const float*)d_in[4];
    const float* k_b   = (const float*)d_in[5];
    const float* v_w   = (const float*)d_in[6];
    const float* v_b   = (const float*)d_in[7];
    const float* gamma = (const float*)d_in[8];
    float* y = (float*)d_out;

    // workspace layout (bytes)
    char* ws = (char*)d_ws;
    half_t* qbuf  = (half_t*)(ws + 0);          //  4 MiB : B*CQ*HW f16
    half_t* kbuf  = (half_t*)(ws + 4194304);    //  4 MiB
    half_t* v2buf = (half_t*)(ws + 8388608);    // 32 MiB : B*C*HW f16
    half_t* v1buf = (half_t*)(ws + 41943040);   // 32 MiB
    float*  ebuf  = (float*) (ws + 75497472);   // 16 MiB : B*HW*128 f32
    half_t* abuf  = (half_t*)(ws + 92274688);   //  8 MiB : B*HW*128 f16

    // 1) projections (f32 -> f16 WMMA GEMMs)
    proj_gemm<<<dim3(32, 1, BB), 256, 0, stream>>>(x2, q_w, q_b, qbuf, CQ);
    proj_gemm<<<dim3(32, 1, BB), 256, 0, stream>>>(x2, k_w, k_b, kbuf, CQ);
    proj_gemm<<<dim3(32, 8, BB), 256, 0, stream>>>(x2, v_w, v_b, v2buf, CC);
    proj_gemm<<<dim3(32, 8, BB), 256, 0, stream>>>(x1, v_w, v_b, v1buf, CC);

    // 2) energies (column then row direction)
    energy_kernel<<<dim3(WW, BB), 256, 0, stream>>>(qbuf, kbuf, ebuf, 0);
    energy_kernel<<<dim3(HH, BB), 256, 0, stream>>>(qbuf, kbuf, ebuf, 1);

    // 3) softmax over 128 logits per pixel
    softmax_kernel<<<dim3(BB * HWP / 8), 256, 0, stream>>>(ebuf, abuf);

    // 4) aggregation: row pass writes y = g*outW + x, column pass adds g*outH
    agg_row_kernel<<<dim3(HH, 4, BB), 256, 0, stream>>>(v2buf, v1buf, abuf, x2, x1, gamma, y);
    agg_col_kernel<<<dim3(WW, 4, BB), 256, 0, stream>>>(v2buf, v1buf, abuf, gamma, y);
}
